// FiLMTransformerDecoderLayer_19026705121924
// MI455X (gfx1250) — matrix operand order repaired and checked by reading, not compiled, not run on hardware
//
#include <hip/hip_runtime.h>

// ---------------------------------------------------------------------------
// FiLM Transformer decoder layer for MI455X (gfx1250, wave32, WMMA).
//  * all GEMMs: v_wmma_f32_16x16x32_f16, 128x128x32 tiles, 8 WMMA/wave/k-step
//  * LDS double-buffered via GLOBAL_LOAD_ASYNC_TO_LDS_B64 (+ s_wait_asynccnt)
//    when the toolchain exposes the builtins; vector-copy fallback otherwise
//  * weights pre-transposed+converted to f16 [N][K]; activations produced in
//    f16 by the upstream kernel (LN / attention / FFN-1 epilogues)
//  * flash attention with trajectory-embedding bias; scores never touch HBM
// ---------------------------------------------------------------------------

typedef _Float16 half_t;
typedef __attribute__((ext_vector_type(16))) _Float16 v16h;
typedef __attribute__((ext_vector_type(4)))  _Float16 v4h;
typedef __attribute__((ext_vector_type(2)))  _Float16 v2h;
typedef __attribute__((ext_vector_type(8)))  float    v8f;
typedef __attribute__((ext_vector_type(2)))  int      v2i;

#define AS_GLOBAL __attribute__((address_space(1)))
#define AS_LOCAL  __attribute__((address_space(3)))

#if defined(__has_builtin)
#if __has_builtin(__builtin_amdgcn_global_load_async_to_lds_b64) && \
    __has_builtin(__builtin_amdgcn_s_wait_asynccnt)
#define ASYNC_LDS 1
#endif
#endif
#ifndef ASYNC_LDS
#define ASYNC_LDS 0
#endif

static __device__ __forceinline__ v8f wmma_f16(v16h a, v16h b, v8f c) {
  // 8 args: (neg_a, A, neg_b, B, c_mod, C, reuse_a, reuse_b)
  return __builtin_amdgcn_wmma_f32_16x16x32_f16(false, a, false, b, (short)0, c,
                                                false, false);
}

// omode for GEMM / LN outputs
#define OM_F32    0  // float  [m][ldc]
#define OM_F16ROW 1  // f16    [m][ldc]
#define OM_QK     2  // f16    [b,h,s,dk]   (attention Q/K layout)
#define OM_V      3  // f16    [b,h,dk,s]   (attention V^T layout)

// ---------------------------------------------------------------------------
// LayerNorm over D=512 per row; one block per row; f32 or f16 output.
// ---------------------------------------------------------------------------
__global__ __launch_bounds__(256) void ln_kernel(
    const float* __restrict__ in, const float* __restrict__ w,
    const float* __restrict__ bv, void* __restrict__ outp, int D, float eps,
    int f16out) {
  const int row = blockIdx.x;
  const float* x = in + (size_t)row * D;
  __shared__ float rs[256];
  __shared__ float rq[256];
  float s = 0.f, s2 = 0.f;
  for (int i = threadIdx.x; i < D; i += 256) {
    float v = x[i];
    s += v;
    s2 += v * v;
  }
  rs[threadIdx.x] = s;
  rq[threadIdx.x] = s2;
  __syncthreads();
  for (int off = 128; off > 0; off >>= 1) {
    if (threadIdx.x < off) {
      rs[threadIdx.x] += rs[threadIdx.x + off];
      rq[threadIdx.x] += rq[threadIdx.x + off];
    }
    __syncthreads();
  }
  float mean = rs[0] / (float)D;
  float var = rq[0] / (float)D - mean * mean;
  float rstd = rsqrtf(var + eps);
  for (int i = threadIdx.x; i < D; i += 256) {
    float v = (x[i] - mean) * rstd * w[i] + bv[i];
    if (f16out)
      ((half_t*)outp)[(size_t)row * D + i] = (half_t)v;
    else
      ((float*)outp)[(size_t)row * D + i] = v;
  }
}

// ---------------------------------------------------------------------------
// Elementwise f32 -> f16 and weight transpose-convert W[K][N] -> Wt[N][K].
// ---------------------------------------------------------------------------
__global__ void cvt16_kernel(const float* __restrict__ src,
                             half_t* __restrict__ dst, size_t n) {
  size_t i = (size_t)blockIdx.x * 256 + threadIdx.x;
  if (i < n) dst[i] = (half_t)src[i];
}

__global__ void cvt_tr_kernel(const float* __restrict__ W,
                              half_t* __restrict__ Wt, int K, int N) {
  size_t i = (size_t)blockIdx.x * 256 + threadIdx.x;
  if (i < (size_t)K * N) {
    int k = (int)(i / (size_t)N);
    int n = (int)(i - (size_t)k * N);
    Wt[(size_t)n * K + k] = (half_t)W[i];
  }
}

// ---------------------------------------------------------------------------
// f16 WMMA GEMM: out = act(A[M,K] @ Wt[N,K]^T + bias), 128x128x32 tiles,
// double-buffered LDS fed by async global->LDS copies.
// M%128==0, N%128==0, K%32==0 at every call site.
// ---------------------------------------------------------------------------
__global__ __launch_bounds__(256) void gemm16_wmma(
    const half_t* __restrict__ A, int lda, const half_t* __restrict__ Wt,
    const float* __restrict__ bias, void* __restrict__ outp, int ldc, int M,
    int N, int K, int relu, int omode) {
  constexpr int LDT = 40;  // padded halves (80B rows; 8B-aligned chunks)
  __shared__ half_t sA[2][128 * LDT];
  __shared__ half_t sB[2][128 * LDT];
  const int t = threadIdx.x;
  const int wave = t >> 5, lane = t & 31;
  const int l = lane & 15, hi = lane >> 4;
  const int wy = wave >> 2;  // 0..1 : 64-row half
  const int wx = wave & 3;   // 0..3 : 32-col quarter
  const int row0 = blockIdx.y * 128, col0 = blockIdx.x * 128;
  const int nk = K >> 5;
  v8f acc[4][2] = {};

  auto load_tile = [&](int buf, int k0) {
    // 128 rows x 32 halves per matrix; 8B chunks -> 1024 lane-ops, 4/thread
#pragma unroll
    for (int j = 0; j < 4; ++j) {
      int chunk = t + j * 256;
      int m = chunk >> 3, c = chunk & 7;
      const half_t* ga = A + (size_t)(row0 + m) * lda + k0 + c * 4;
      const half_t* gb = Wt + (size_t)(col0 + m) * K + k0 + c * 4;
      half_t* la = &sA[buf][m * LDT + c * 4];
      half_t* lb = &sB[buf][m * LDT + c * 4];
#if ASYNC_LDS
      __builtin_amdgcn_global_load_async_to_lds_b64((AS_GLOBAL v2i*)ga,
                                                    (AS_LOCAL v2i*)la, 0, 0);
      __builtin_amdgcn_global_load_async_to_lds_b64((AS_GLOBAL v2i*)gb,
                                                    (AS_LOCAL v2i*)lb, 0, 0);
#else
      *(v4h*)la = *(const v4h*)ga;
      *(v4h*)lb = *(const v4h*)gb;
#endif
    }
  };

  load_tile(0, 0);
  for (int i = 0; i < nk; ++i) {
    const int cur = i & 1;
#if ASYNC_LDS
    __builtin_amdgcn_s_wait_asynccnt(0);
#endif
    __syncthreads();
    if (i + 1 < nk) load_tile(cur ^ 1, (i + 1) * 32);

    const half_t* pa = &sA[cur][0];
    const half_t* pb = &sB[cur][0];
    v16h afr[4], bfr[2];
#pragma unroll
    for (int tf = 0; tf < 4; ++tf) {
      int row = wy * 64 + tf * 16 + l;
#pragma unroll
      for (int g = 0; g < 8; ++g) {
        int kb = (hi ? 16 : 0) + 2 * g;
        v2h p = *(const v2h*)&pa[row * LDT + kb];
        afr[tf][2 * g] = p[0];
        afr[tf][2 * g + 1] = p[1];
      }
    }
#pragma unroll
    for (int sf = 0; sf < 2; ++sf) {
      int col = wx * 32 + sf * 16 + l;
#pragma unroll
      for (int g = 0; g < 8; ++g) {
        int kk = (hi ? 16 : 0) + 2 * g;
        v2h p = *(const v2h*)&pb[col * LDT + kk];
        bfr[sf][2 * g] = p[0];
        bfr[sf][2 * g + 1] = p[1];
      }
    }
#pragma unroll
    for (int tf = 0; tf < 4; ++tf)
#pragma unroll
      for (int sf = 0; sf < 2; ++sf)
        acc[tf][sf] = wmma_f16(afr[tf], bfr[sf], acc[tf][sf]);
  }

  // epilogue
  const int mb = row0 + wy * 64;
  const int nb = col0 + wx * 32;
#pragma unroll
  for (int sf = 0; sf < 2; ++sf) {
    const int n = nb + sf * 16 + l;
    const float bia = bias ? bias[n] : 0.f;
#pragma unroll
    for (int tf = 0; tf < 4; ++tf) {
#pragma unroll
      for (int r = 0; r < 8; ++r) {
        int m = mb + tf * 16 + (hi ? 8 : 0) + r;
        float v = acc[tf][sf][r] + bia;
        if (relu) v = fmaxf(v, 0.f);
        if (omode == OM_F32) {
          ((float*)outp)[(size_t)m * ldc + n] = v;
        } else if (omode == OM_F16ROW) {
          ((half_t*)outp)[(size_t)m * ldc + n] = (half_t)v;
        } else {
          int bb = m >> 10, s = m & 1023;
          int h = n >> 6, dk = n & 63;
          size_t idx =
              (omode == OM_QK)
                  ? ((((size_t)bb * 8 + h) * 1024 + s) * 64 + dk)
                  : ((((size_t)bb * 8 + h) * 64 + dk) * 1024 + s);
          ((half_t*)outp)[idx] = (half_t)v;
        }
      }
    }
  }
}

// ---------------------------------------------------------------------------
// Flash attention with trajectory-embedding bias.
// One wave per (b, h, 16-query tile). S=1024, DK=64, H=8, NEMB=10.
// scores = (q.k)/8 + indexed[q][trj_dist[b,q,k]],  indexed = q . emb[n,h,:]
// Output written as f16 row-major [token][512] (feeds the fc GEMM).
// ---------------------------------------------------------------------------
__global__ __launch_bounds__(256) void attn_kernel(
    const half_t* __restrict__ Qh, const half_t* __restrict__ Kh,
    const half_t* __restrict__ Vt, const float* __restrict__ emb,
    const int* __restrict__ dist, half_t* __restrict__ outp) {
  __shared__ float idxb[8][160];   // per-wave indexed[q:16][n:10]
  __shared__ half_t pbuf[8][512];  // per-wave P tile [16][32]
  const int wave = threadIdx.x >> 5, lane = threadIdx.x & 31;
  const int l = lane & 15, hi = lane >> 4;
  const int task = blockIdx.x * 8 + wave;
  const int qt = task & 63;
  const int h = (task >> 6) & 7;
  const int bb = task >> 9;
  const int q0 = qt * 16;
  const half_t* Qp = Qh + ((size_t)(bb * 8 + h)) * 1024 * 64;
  const half_t* Kp = Kh + ((size_t)(bb * 8 + h)) * 1024 * 64;
  const half_t* Vp = Vt + ((size_t)(bb * 8 + h)) * 64 * 1024;

  // Q A-fragments (d=0..31 and d=32..63) -- unscaled q.
  v16h qa0, qa1;
#pragma unroll
  for (int g = 0; g < 8; ++g) {
    int kb = (hi ? 16 : 0) + 2 * g;
    v2h p0 = *(const v2h*)&Qp[(size_t)(q0 + l) * 64 + kb];
    v2h p1 = *(const v2h*)&Qp[(size_t)(q0 + l) * 64 + 32 + kb];
    qa0[2 * g] = p0[0];
    qa0[2 * g + 1] = p0[1];
    qa1[2 * g] = p1[0];
    qa1[2 * g + 1] = p1[1];
  }
  // indexed[q][n] = q . emb[n, h*64 : h*64+64]
  float* ib = idxb[wave];
  for (int tt = lane; tt < 160; tt += 32) {
    int qq = tt / 10, n = tt - qq * 10;
    const half_t* qr = Qp + (size_t)(q0 + qq) * 64;
    const float* er = emb + n * 512 + h * 64;
    float s = 0.f;
    for (int d = 0; d < 64; ++d) s += (float)qr[d] * er[d];
    ib[tt] = s;
  }
  half_t* P = pbuf[wave];
  float rowm[8], rowl[8];
  v8f acc0 = {}, acc1 = {}, acc2 = {}, acc3 = {};
#pragma unroll
  for (int r = 0; r < 8; ++r) {
    rowm[r] = -3.0e38f;
    rowl[r] = 0.f;
  }
  const float invt = 0.125f;  // 1/sqrt(DK)
  const int* drow = dist + ((size_t)bb * 1024 + q0) * 1024;

  for (int kb0 = 0; kb0 < 1024; kb0 += 32) {
    v8f s0 = {}, s1 = {};
    v16h bl0, bh0, bl1, bh1;
#pragma unroll
    for (int g = 0; g < 8; ++g) {
      int kk = (hi ? 16 : 0) + 2 * g;
      size_t r0 = (size_t)(kb0 + l) * 64;
      size_t r1 = (size_t)(kb0 + 16 + l) * 64;
      v2h pa = *(const v2h*)&Kp[r0 + kk];
      v2h pb = *(const v2h*)&Kp[r0 + 32 + kk];
      v2h pc = *(const v2h*)&Kp[r1 + kk];
      v2h pd = *(const v2h*)&Kp[r1 + 32 + kk];
      bl0[2 * g] = pa[0]; bl0[2 * g + 1] = pa[1];
      bh0[2 * g] = pb[0]; bh0[2 * g + 1] = pb[1];
      bl1[2 * g] = pc[0]; bl1[2 * g + 1] = pc[1];
      bh1[2 * g] = pd[0]; bh1[2 * g + 1] = pd[1];
    }
    s0 = wmma_f16(qa0, bl0, s0);
    s0 = wmma_f16(qa1, bh0, s0);
    s1 = wmma_f16(qa0, bl1, s1);
    s1 = wmma_f16(qa1, bh1, s1);

    float alpha[8];
#pragma unroll
    for (int r = 0; r < 8; ++r) {
      int row = (hi ? 8 : 0) + r;
      const int* dr = drow + (size_t)row * 1024 + kb0;
      int i0 = dr[l];
      int i1 = dr[16 + l];
      float v0 = s0[r] * invt + ib[row * 10 + i0];
      float v1 = s1[r] * invt + ib[row * 10 + i1];
      float m = fmaxf(v0, v1);
      m = fmaxf(m, __shfl_xor(m, 1, 32));
      m = fmaxf(m, __shfl_xor(m, 2, 32));
      m = fmaxf(m, __shfl_xor(m, 4, 32));
      m = fmaxf(m, __shfl_xor(m, 8, 32));
      float nm = fmaxf(rowm[r], m);
      alpha[r] = __expf(rowm[r] - nm);
      rowm[r] = nm;
      float p0 = __expf(v0 - nm);
      float p1 = __expf(v1 - nm);
      P[row * 32 + l] = (half_t)p0;
      P[row * 32 + 16 + l] = (half_t)p1;
      float sm = p0 + p1;
      sm += __shfl_xor(sm, 1, 32);
      sm += __shfl_xor(sm, 2, 32);
      sm += __shfl_xor(sm, 4, 32);
      sm += __shfl_xor(sm, 8, 32);
      rowl[r] = rowl[r] * alpha[r] + sm;
    }
#pragma unroll
    for (int r = 0; r < 8; ++r) {
      acc0[r] *= alpha[r];
      acc1[r] *= alpha[r];
      acc2[r] *= alpha[r];
      acc3[r] *= alpha[r];
    }
    v16h pa;
#pragma unroll
    for (int g = 0; g < 8; ++g) {
      int kb = (hi ? 16 : 0) + 2 * g;
      v2h pp = *(const v2h*)&P[l * 32 + kb];
      pa[2 * g] = pp[0];
      pa[2 * g + 1] = pp[1];
    }
    v16h vb0, vb1, vb2, vb3;
#pragma unroll
    for (int g = 0; g < 8; ++g) {
      int kk = (hi ? 16 : 0) + 2 * g;
      v2h a0 = *(const v2h*)&Vp[(size_t)(l)*1024 + kb0 + kk];
      v2h a1 = *(const v2h*)&Vp[(size_t)(16 + l) * 1024 + kb0 + kk];
      v2h a2 = *(const v2h*)&Vp[(size_t)(32 + l) * 1024 + kb0 + kk];
      v2h a3 = *(const v2h*)&Vp[(size_t)(48 + l) * 1024 + kb0 + kk];
      vb0[2 * g] = a0[0]; vb0[2 * g + 1] = a0[1];
      vb1[2 * g] = a1[0]; vb1[2 * g + 1] = a1[1];
      vb2[2 * g] = a2[0]; vb2[2 * g + 1] = a2[1];
      vb3[2 * g] = a3[0]; vb3[2 * g + 1] = a3[1];
    }
    acc0 = wmma_f16(pa, vb0, acc0);
    acc1 = wmma_f16(pa, vb1, acc1);
    acc2 = wmma_f16(pa, vb2, acc2);
    acc3 = wmma_f16(pa, vb3, acc3);
  }
#pragma unroll
  for (int r = 0; r < 8; ++r) {
    int row = (hi ? 8 : 0) + r;
    float inv = 1.0f / rowl[r];
    size_t o = (((size_t)bb * 1024) + (q0 + row)) * 512 + h * 64 + l;
    outp[o + 0] = (half_t)(acc0[r] * inv);
    outp[o + 16] = (half_t)(acc1[r] * inv);
    outp[o + 32] = (half_t)(acc2[r] * inv);
    outp[o + 48] = (half_t)(acc3[r] * inv);
  }
}

// ---------------------------------------------------------------------------
// FiLM helpers: mish(t), the tiny (4 x 1024) FiLM GEMM, then combine.
// ---------------------------------------------------------------------------
__global__ void mish_kernel(const float* __restrict__ t, float* __restrict__ mt,
                            int n) {
  int i = blockIdx.x * 256 + threadIdx.x;
  if (i < n) {
    float x = t[i];
    float sp = (x > 20.f) ? x : log1pf(expf(x));
    mt[i] = x * tanhf(sp);
  }
}

__global__ void film_gemm(const float* __restrict__ mt,
                          const float* __restrict__ W,
                          const float* __restrict__ bias,
                          float* __restrict__ film, int K, int NO) {
  int idx = blockIdx.x * 256 + threadIdx.x;  // B*NO threads
  int bb = idx / NO, j = idx - bb * NO;
  if (bb < 4) {
    float s = bias[j];
    for (int k = 0; k < K; ++k) s += mt[bb * K + k] * W[(size_t)k * NO + j];
    film[idx] = s;
  }
}

// out = base + (scale+1)*branch + shift; film[b][0:512]=scale,[512:1024]=shift
__global__ void film_combine(const float* __restrict__ base,
                             const float* __restrict__ br,
                             const float* __restrict__ film,
                             float* __restrict__ outp) {
  size_t i = (size_t)blockIdx.x * 256 + threadIdx.x;  // < 4096*512
  int bb = (int)(i >> 19);
  int d = (int)(i & 511);
  float sc = film[bb * 1024 + d];
  float sh = film[bb * 1024 + 512 + d];
  outp[i] = base[i] + (sc + 1.f) * br[i] + sh;
}

// ---------------------------------------------------------------------------
// Host-side orchestration.
// Input order (recursive dict-insertion flatten of setup_inputs):
//  0 tgt, 1 memory, 2 t, 3 traj_emb(unused), 4 trj_dist,
//  5..10  sa{wq,wk,wv,fc,ln_w,ln_b}, 11..16 ca{...}, 17 emb,
//  18/19 lin1_w/b, 20/21 lin2_w/b, 22/23 lin3_w/b,
//  24..31 norm1..4 {w,b}, 32..37 film1..3 {w,b}, 38..52 traj (unused)
// ---------------------------------------------------------------------------
extern "C" void kernel_launch(void* const* d_in, const int* in_sizes, int n_in,
                              void* d_out, int out_size, void* d_ws,
                              size_t ws_size, hipStream_t stream) {
  (void)in_sizes; (void)n_in; (void)out_size; (void)ws_size;
  const float* tgt    = (const float*)d_in[0];
  const float* memory = (const float*)d_in[1];
  const float* t_in   = (const float*)d_in[2];
  const int*   dist   = (const int*)d_in[4];
  const float* sa_wq  = (const float*)d_in[5];
  const float* sa_wk  = (const float*)d_in[6];
  const float* sa_wv  = (const float*)d_in[7];
  const float* sa_fc  = (const float*)d_in[8];
  const float* sa_lnw = (const float*)d_in[9];
  const float* sa_lnb = (const float*)d_in[10];
  const float* ca_wq  = (const float*)d_in[11];
  const float* ca_wk  = (const float*)d_in[12];
  const float* ca_wv  = (const float*)d_in[13];
  const float* ca_fc  = (const float*)d_in[14];
  const float* ca_lnw = (const float*)d_in[15];
  const float* ca_lnb = (const float*)d_in[16];
  const float* emb    = (const float*)d_in[17];
  const float* lin1_w = (const float*)d_in[18];
  const float* lin1_b = (const float*)d_in[19];
  const float* lin2_w = (const float*)d_in[20];
  const float* lin2_b = (const float*)d_in[21];
  const float* lin3_w = (const float*)d_in[22];
  const float* lin3_b = (const float*)d_in[23];
  const float* n1w = (const float*)d_in[24];
  const float* n1b = (const float*)d_in[25];
  const float* n2w = (const float*)d_in[26];
  const float* n2b = (const float*)d_in[27];
  const float* n3w = (const float*)d_in[28];
  const float* n3b = (const float*)d_in[29];
  const float* n4w = (const float*)d_in[30];
  const float* n4b = (const float*)d_in[31];
  const float* f1w = (const float*)d_in[32];
  const float* f1b = (const float*)d_in[33];
  const float* f2w = (const float*)d_in[34];
  const float* f2b = (const float*)d_in[35];
  const float* f3w = (const float*)d_in[36];
  const float* f3b = (const float*)d_in[37];

  const size_t ND = 4096ull * 512;
  char* base = (char*)d_ws;
  float* xcur = (float*)base;  base += ND * 4;
  float* xn   = (float*)base;  base += ND * 4;
  float* x1   = (float*)base;  base += ND * 4;
  half_t* a16 = (half_t*)base; base += ND * 2;                 // act f16 [m][512]
  half_t* h16 = (half_t*)base; base += 4096ull * 2048 * 2;     // FFN hidden f16
  half_t* w16 = (half_t*)base; base += 2048ull * 512 * 2;      // Wt f16 [N][K]
  half_t* qh  = (half_t*)base; base += ND * 2;
  half_t* kh  = (half_t*)base; base += ND * 2;
  half_t* vt  = (half_t*)base; base += ND * 2;
  float* mish_t = (float*)base; base += 2048 * 4;
  float* film   = (float*)base; base += 4096 * 4;

  dim3 blk(256);
  auto cvtw = [&](const float* W, int K, int N) {
    cvt_tr_kernel<<<(K * N + 255) / 256, blk, 0, stream>>>(W, w16, K, N);
  };
  auto gemm = [&](const half_t* A, int lda, const float* bias, void* C,
                  int ldc, int M, int N, int K, int relu, int omode) {
    gemm16_wmma<<<dim3(N / 128, M / 128), blk, 0, stream>>>(
        A, lda, w16, bias, C, ldc, M, N, K, relu, omode);
  };

  // ---- block 1: self-attention + FiLM ----
  ln_kernel<<<4096, blk, 0, stream>>>(tgt, n1w, n1b, a16, 512, 1e-5f, 1);
  cvtw(sa_wq, 512, 512); gemm(a16, 512, nullptr, qh, 0, 4096, 512, 512, 0, OM_QK);
  cvtw(sa_wk, 512, 512); gemm(a16, 512, nullptr, kh, 0, 4096, 512, 512, 0, OM_QK);
  cvtw(sa_wv, 512, 512); gemm(a16, 512, nullptr, vt, 0, 4096, 512, 512, 0, OM_V);
  mish_kernel<<<8, blk, 0, stream>>>(t_in, mish_t, 2048);
  film_gemm<<<16, blk, 0, stream>>>(mish_t, f1w, f1b, film, 512, 1024);
  attn_kernel<<<256, blk, 0, stream>>>(qh, kh, vt, emb, dist, a16);
  cvtw(sa_fc, 512, 512); gemm(a16, 512, nullptr, xn, 512, 4096, 512, 512, 0, OM_F32);
  ln_kernel<<<4096, blk, 0, stream>>>(xn, sa_lnw, sa_lnb, x1, 512, 1e-6f, 0);
  film_combine<<<8192, blk, 0, stream>>>(tgt, x1, film, xcur);

  // ---- block 2: cross-attention + FiLM ----
  ln_kernel<<<4096, blk, 0, stream>>>(xcur, n2w, n2b, a16, 512, 1e-5f, 1);
  cvtw(ca_wq, 512, 512); gemm(a16, 512, nullptr, qh, 0, 4096, 512, 512, 0, OM_QK);
  cvt16_kernel<<<8192, blk, 0, stream>>>(memory, a16, ND);
  cvtw(ca_wk, 512, 512); gemm(a16, 512, nullptr, kh, 0, 4096, 512, 512, 0, OM_QK);
  cvtw(ca_wv, 512, 512); gemm(a16, 512, nullptr, vt, 0, 4096, 512, 512, 0, OM_V);
  film_gemm<<<16, blk, 0, stream>>>(mish_t, f2w, f2b, film, 512, 1024);
  attn_kernel<<<256, blk, 0, stream>>>(qh, kh, vt, emb, dist, a16);
  cvtw(ca_fc, 512, 512); gemm(a16, 512, nullptr, xn, 512, 4096, 512, 512, 0, OM_F32);
  ln_kernel<<<4096, blk, 0, stream>>>(xn, ca_lnw, ca_lnb, x1, 512, 1e-6f, 0);
  film_combine<<<8192, blk, 0, stream>>>(xcur, x1, film, xcur);

  // ---- block 3: FFN + FiLM ----
  ln_kernel<<<4096, blk, 0, stream>>>(xcur, n3w, n3b, a16, 512, 1e-5f, 1);
  cvtw(lin1_w, 512, 2048);
  gemm(a16, 512, lin1_b, h16, 2048, 4096, 2048, 512, 1, OM_F16ROW);
  cvtw(lin2_w, 2048, 512);
  gemm(h16, 2048, lin2_b, x1, 512, 4096, 512, 2048, 0, OM_F32);
  film_gemm<<<16, blk, 0, stream>>>(mish_t, f3w, f3b, film, 512, 1024);
  film_combine<<<8192, blk, 0, stream>>>(xcur, x1, film, xcur);

  // ---- output projection ----
  ln_kernel<<<4096, blk, 0, stream>>>(xcur, n4w, n4b, a16, 512, 1e-5f, 1);
  cvtw(lin3_w, 512, 512);
  gemm(a16, 512, lin3_b, (float*)d_out, 512, 4096, 512, 512, 0, OM_F32);
}
